// Model_52381421142388
// MI455X (gfx1250) — compile-verified, matrix operand order checked by reading, not run on hardware
//
#include <hip/hip_runtime.h>

#define TT 7
#define BB 2048
#define NN 90
#define FF 27
#define HH 30

// LDS strides (f16 elements)
#define XS 36    // row stride for sXc/sF/sL (96 rows)
#define AS 96    // row stride for sAdj
#define TS 100   // row stride for transposed lg/s1g  [n][k]
#define ES 292   // row stride for transposed im2col  [n][k]

typedef __attribute__((ext_vector_type(16))) _Float16 v16h;
typedef __attribute__((ext_vector_type(8)))  float    v8f;

__device__ __forceinline__ v8f wmma_f16(v16h a, v16h b, v8f c) {
  // v_wmma_f32_16x16x32_f16: (neg_a, A, neg_b, B, c_mod, C, reuse_a, reuse_b)
  return __builtin_amdgcn_wmma_f32_16x16x32_f16(false, a, false, b, (short)0, c, false, false);
}

// A/B fragment loader (16-bit 16x32 layout).
// lanes 0-15: K in {0..7,16..23}; lanes 16-31: K in {8..15,24..31}; P = lane&15.
template <class G>
__device__ __forceinline__ v16h frag16(G get, int p0, int k0) {
  int lane = threadIdx.x & 31;
  int p = p0 + (lane & 15);
  int h = (lane >> 4) & 1;
  v16h r;
#pragma unroll
  for (int j = 0; j < 8; ++j) {
    int kk = ((j & 4) << 2) + h * 8 + (j & 3) * 2;
    r[2 * j]     = get(p, k0 + kk);
    r[2 * j + 1] = get(p, k0 + kk + 1);
  }
  return r;
}

// C/D layout: VGPR v -> M = v (lanes 0-15) / v+8 (lanes 16-31), N = lane&15.
template <class S>
__device__ __forceinline__ void storeC(const v8f& c, S put, int m0, int n0) {
  int lane = threadIdx.x & 31;
  int n = n0 + (lane & 15);
  int mb = m0 + ((lane >> 4) << 3);
#pragma unroll
  for (int v = 0; v < 8; ++v) put(mb + v, n, c[v]);
}

__device__ __forceinline__ float wave_sum(float v) {
#pragma unroll
  for (int off = 16; off > 0; off >>= 1) v += __shfl_down(v, off, 32);
  return v;
}

// ---------------- uncertainty: sig[r], r = t*B+b (t-major flat) ----------------
__global__ __launch_bounds__(128)
void uncert_kernel(const float* __restrict__ fdata, const float* __restrict__ unw,
                   const float* __restrict__ unb, float* __restrict__ sig) {
  int r = blockIdx.x * 4 + (threadIdx.x >> 5);
  int lane = threadIdx.x & 31;
  int t = r / BB, b = r % BB;
  const float* src = fdata + ((size_t)b * TT + t) * (NN * FF);
  float s = 0.f;
  for (int i = lane; i < NN * FF; i += 32) s += src[i] * unw[i];
  s = wave_sum(s);
  if (lane == 0) sig[r] = 1.0f / (1.0f + expf(-(s + unb[0])));
}

// ---------------- temporal attention + scrambled reshape into seq ----------------
__global__ __launch_bounds__(128)
void tatt_kernel(const float* __restrict__ fdata, const float* __restrict__ U1,
                 const float* __restrict__ U2, const float* __restrict__ U3,
                 const float* __restrict__ b_e, const float* __restrict__ v_e,
                 float* __restrict__ seq) {
  __shared__ _Float16 sfd[TT * NN * FF];
  __shared__ float sl1[TT * FF];
  __shared__ float sRhs[NN * TT];
  __shared__ float sL2[TT * NN];
  __shared__ float sP[TT * TT];
  __shared__ float sS[TT * TT];
  __shared__ float sTt[TT * TT];
  int b = blockIdx.x, tid = threadIdx.x;
  const float* src = fdata + (size_t)b * (TT * NN * FF);
  for (int i = tid; i < TT * NN * FF; i += 128) sfd[i] = (_Float16)src[i];
  __syncthreads();
  for (int i = tid; i < TT * FF; i += 128) {
    int tq = i / FF, f = i % FF;
    float s = 0.f;
    for (int v = 0; v < NN; ++v) s += (float)sfd[tq * (NN * FF) + v * FF + f] * U1[v];
    sl1[i] = s;
  }
  for (int i = tid; i < NN * TT; i += 128) {
    int v = i / TT, tq = i % TT;
    float s = 0.f;
    for (int f = 0; f < FF; ++f) s += (float)sfd[tq * (NN * FF) + v * FF + f] * U3[f];
    sRhs[i] = s;
  }
  __syncthreads();
  for (int i = tid; i < TT * NN; i += 128) {
    int tq = i / NN, n = i % NN;
    float s = 0.f;
    for (int f = 0; f < FF; ++f) s += sl1[tq * FF + f] * U2[f * NN + n];
    sL2[i] = s;
  }
  __syncthreads();
  for (int i = tid; i < TT * TT; i += 128) {
    int tq = i / TT, sq = i % TT;
    float s = 0.f;
    for (int v = 0; v < NN; ++v) s += sL2[tq * NN + v] * sRhs[v * TT + sq];
    sP[i] = 1.0f / (1.0f + expf(-(s + b_e[i])));
  }
  __syncthreads();
  for (int i = tid; i < TT * TT; i += 128) {
    int kq = i / TT, iq = i % TT;
    float s = 0.f;
    for (int j = 0; j < TT; ++j) s += v_e[iq * TT + j] * sP[kq * TT + j];
    sS[kq * TT + iq] = s;
  }
  __syncthreads();
  if (tid < TT) {
    int iq = tid;
    float m = -1e30f;
    for (int k = 0; k < TT; ++k) m = fmaxf(m, sS[k * TT + iq]);
    float den = 0.f;
    for (int k = 0; k < TT; ++k) den += expf(sS[k * TT + iq] - m);
    for (int k = 0; k < TT; ++k) sTt[k * TT + iq] = expf(sS[k * TT + iq] - m) / den;
  }
  __syncthreads();
  // y[i2,j] = sum_s fp[i2,s]*tatt[s,j]; flat idx reinterpreted as (t,n,f)
  for (int i = tid; i < NN * FF * TT; i += 128) {
    int i2 = i / TT, j = i % TT;
    float s = 0.f;
#pragma unroll
    for (int sI = 0; sI < TT; ++sI) s += (float)sfd[sI * (NN * FF) + i2] * sTt[sI * TT + j];
    int tD = i / (NN * FF), rem = i % (NN * FF);
    seq[(size_t)tD * ((size_t)BB * NN * FF) + (size_t)b * (NN * FF) + rem] = s;
  }
}

// ---------------- init: zero accum, build f16 weights (gc1/gc2 transposed) ----------------
__global__ void init_kernel(float* finalA, float* gmAcc, _Float16* Wf, _Float16* G1t,
                            _Float16* G2t, const float* cw, const float* gc1, const float* gc2) {
  int gid = blockIdx.x * blockDim.x + threadIdx.x;
  int stride = gridDim.x * blockDim.x;
  for (int i = gid; i < BB * NN * HH; i += stride) finalA[i] = 0.f;
  for (int i = gid; i < 96 * 288; i += stride) {
    int o = i / 288, c = i % 288;
    Wf[i] = (o < NN && c < 270) ? (_Float16)cw[o * 270 + c] : (_Float16)0.f;
  }
  for (int i = gid; i < 1024; i += stride) {
    int n = i >> 5, k = i & 31;  // transposed: row n, col k
    G1t[i] = (k < FF && n < HH) ? (_Float16)gc1[k * HH + n] : (_Float16)0.f;
    G2t[i] = (k < HH && n < HH) ? (_Float16)gc2[k * HH + n] : (_Float16)0.f;
  }
  if (gid < 32) gmAcc[gid] = 0.f;
}

// ---------------- gl[f] = lin_b[f] + lin_w @ mean(g_prev); then reset accum ----------------
__global__ void gl_kernel(float* gmAcc, float* gl, const float* lw, const float* lb) {
  int tid = threadIdx.x;
  float g = 0.f;
  if (tid < FF) {
    for (int h = 0; h < HH; ++h) g += lw[tid * HH + h] * gmAcc[h];
    g = lb[tid] + g * (1.0f / (float)(BB * NN));
  }
  __syncthreads();
  if (tid < FF) gl[tid] = g;
  if (tid < HH) gmAcc[tid] = 0.f;
}

// ---------------- ST block: one workgroup (4 waves) per batch element ----------------
__global__ __launch_bounds__(128)
void st_kernel(const float* __restrict__ seqT, const float* __restrict__ sig,
               const float* __restrict__ gl, const float* __restrict__ cnnb,
               const _Float16* __restrict__ Wf, const _Float16* __restrict__ G1t,
               const _Float16* __restrict__ G2t, float* __restrict__ finalA,
               float* __restrict__ gmAcc, int t) {
  // time-multiplexed LDS arena (64768 bytes total)
  __shared__ __align__(16) unsigned char smem[64768];
  float*    sFf   = (float*)smem;                       // 96x32 f32 staging (12288 B)
  _Float16* fexpT = (_Float16*)smem;                    // 32x292 f16 im2col  (18688 B)
  _Float16* sXc   = (_Float16*)(smem + 18688);          // 96xXS  (6912 B)
  _Float16* sF    = (_Float16*)(smem + 18688 + 6912);   // 96xXS  (6912 B)
  _Float16* sAdj  = (_Float16*)(smem + 18688 + 13824);  // 96x96  (18432 B)
  _Float16* sL    = (_Float16*)(smem + 18688 + 32256);  // 96xXS  (6912 B) l/s1/g_out
  _Float16* sTmT  = (_Float16*)(smem + 18688 + 39168);  // 32xTS  (6400 B) lg^T / s1g^T
  float*    sD    = (float*)(smem + 18688 + 45568);     // 96 f32 (384 B)
  float*    sRed  = (float*)(smem + 18688 + 45952);     // 32 f32 (128 B)

  const int tid = threadIdx.x;
  const int wave = tid >> 5;
  const int b = blockIdx.x;
  const float* fsrc = seqT + (size_t)b * (NN * FF);

  for (int i = tid; i < NN * FF; i += 128) sFf[(i / FF) * 32 + (i % FF)] = fsrc[i];
  __syncthreads();
  for (int i = tid; i < 96; i += 128) {
    if (i < NN) {
      float s = 0.f;
      for (int c = 0; c < FF; ++c) s += sFf[i * 32 + c];
      float mn = s * (1.0f / FF);
      for (int c = 0; c < 32; ++c) {
        float v = (c < FF) ? sFf[i * 32 + c] : 0.f;
        sF[i * XS + c] = (_Float16)v;
        sXc[i * XS + c] = (_Float16)((c < FF) ? (v - mn) : 0.f);
      }
    } else {
      for (int c = 0; c < 32; ++c) { sF[i * XS + c] = (_Float16)0.f; sXc[i * XS + c] = (_Float16)0.f; }
    }
  }
  __syncthreads();

  // cov = xc @ xc^T  -> sAdj (f16), 36 tiles
  for (int tt2 = wave * 9; tt2 < wave * 9 + 9; ++tt2) {
    int mt = tt2 / 6, nt = tt2 % 6;
    v8f acc = {0.f, 0.f, 0.f, 0.f, 0.f, 0.f, 0.f, 0.f};
    v16h a = frag16([&](int m, int k) { return sXc[m * XS + k]; }, mt * 16, 0);
    v16h bf = frag16([&](int n, int k) { return sXc[n * XS + k]; }, nt * 16, 0);
    acc = wmma_f16(a, bf, acc);
    storeC(acc, [&](int m, int n, float v) { sAdj[m * AS + n] = (_Float16)v; }, mt * 16, nt * 16);
  }
  __syncthreads();
  for (int i = tid; i < 96; i += 128)
    sD[i] = (i < NN) ? sqrtf(fmaxf((float)sAdj[i * AS + i], 1e-12f)) : 1.0f;
  __syncthreads();
  for (int i = tid; i < 96 * 96; i += 128) {
    int m = i / 96, n = i % 96;
    float v = (m < NN && n < NN) ? ((float)sAdj[m * AS + n] / (sD[m] * sD[n])) : 0.f;
    sAdj[m * AS + n] = (_Float16)v;
  }
  __syncthreads();

  // im2col of f (transposed [n][k]) -- sXc/sFf dead, bufA reused
  for (int i = tid; i < 32 * 288; i += 128) {
    int n = i / 288, k = i % 288;
    int i2 = k / 3, r = k - i2 * 3, x = n + r - 1;
    fexpT[n * ES + k] =
        (i2 < NN && n < FF && (unsigned)x < (unsigned)FF) ? sF[i2 * XS + x] : (_Float16)0.f;
  }
  __syncthreads();

  // l = conv1x3(f) + cnnb + gl  (96x288 @ 288x32) -> sL
  for (int tt2 = wave * 3; tt2 < wave * 3 + 3; ++tt2) {
    int mt = tt2 >> 1, nt = tt2 & 1;
    v8f acc = {0.f, 0.f, 0.f, 0.f, 0.f, 0.f, 0.f, 0.f};
    for (int k0 = 0; k0 < 288; k0 += 32) {
      v16h a = frag16([&](int m, int k) { return Wf[m * 288 + k]; }, mt * 16, k0);
      v16h bf = frag16([&](int n, int k) { return fexpT[n * ES + k]; }, nt * 16, k0);
      acc = wmma_f16(a, bf, acc);
    }
    storeC(acc, [&](int m, int n, float v) {
      float o = (m < NN && n < FF) ? (v + cnnb[m] + gl[n]) : 0.f;
      sL[m * XS + n] = (_Float16)o;
    }, mt * 16, nt * 16);
  }
  __syncthreads();

  // lg = l @ gc1 -> sTmT (transposed)
  for (int tt2 = wave * 3; tt2 < wave * 3 + 3; ++tt2) {
    int mt = tt2 >> 1, nt = tt2 & 1;
    v8f acc = {0.f, 0.f, 0.f, 0.f, 0.f, 0.f, 0.f, 0.f};
    v16h a = frag16([&](int m, int k) { return sL[m * XS + k]; }, mt * 16, 0);
    v16h bf = frag16([&](int n, int k) { return G1t[n * 32 + k]; }, nt * 16, 0);
    acc = wmma_f16(a, bf, acc);
    storeC(acc, [&](int m, int n, float v) { sTmT[n * TS + m] = (_Float16)v; }, mt * 16, nt * 16);
  }
  __syncthreads();

  // s1 = adj @ lg -> sL
  for (int tt2 = wave * 3; tt2 < wave * 3 + 3; ++tt2) {
    int mt = tt2 >> 1, nt = tt2 & 1;
    v8f acc = {0.f, 0.f, 0.f, 0.f, 0.f, 0.f, 0.f, 0.f};
    for (int k0 = 0; k0 < 96; k0 += 32) {
      v16h a = frag16([&](int m, int k) { return sAdj[m * AS + k]; }, mt * 16, k0);
      v16h bf = frag16([&](int n, int k) { return sTmT[n * TS + k]; }, nt * 16, k0);
      acc = wmma_f16(a, bf, acc);
    }
    storeC(acc, [&](int m, int n, float v) { sL[m * XS + n] = (_Float16)v; }, mt * 16, nt * 16);
  }
  __syncthreads();

  // s1g = s1 @ gc2 -> sTmT (transposed)
  for (int tt2 = wave * 3; tt2 < wave * 3 + 3; ++tt2) {
    int mt = tt2 >> 1, nt = tt2 & 1;
    v8f acc = {0.f, 0.f, 0.f, 0.f, 0.f, 0.f, 0.f, 0.f};
    v16h a = frag16([&](int m, int k) { return sL[m * XS + k]; }, mt * 16, 0);
    v16h bf = frag16([&](int n, int k) { return G2t[n * 32 + k]; }, nt * 16, 0);
    acc = wmma_f16(a, bf, acc);
    storeC(acc, [&](int m, int n, float v) { sTmT[n * TS + m] = (_Float16)v; }, mt * 16, nt * 16);
  }
  if (tid < 32) sRed[tid] = 0.f;
  __syncthreads();

  // g_out = adj @ s1g -> sL (+ column sums for next-iteration gm)
  for (int tt2 = wave * 3; tt2 < wave * 3 + 3; ++tt2) {
    int mt = tt2 >> 1, nt = tt2 & 1;
    v8f acc = {0.f, 0.f, 0.f, 0.f, 0.f, 0.f, 0.f, 0.f};
    for (int k0 = 0; k0 < 96; k0 += 32) {
      v16h a = frag16([&](int m, int k) { return sAdj[m * AS + k]; }, mt * 16, k0);
      v16h bf = frag16([&](int n, int k) { return sTmT[n * TS + k]; }, nt * 16, k0);
      acc = wmma_f16(a, bf, acc);
    }
    storeC(acc, [&](int m, int n, float v) {
      if (m < NN && n < HH) { sL[m * XS + n] = (_Float16)v; atomicAdd(&sRed[n], v); }
      else sL[m * XS + n] = (_Float16)0.f;
    }, mt * 16, nt * 16);
  }
  __syncthreads();
  if (tid < HH) atomicAdd(&gmAcc[tid], sRed[tid]);

  // im2col of g_out (transposed [n][k]) into bufA
  for (int i = tid; i < 32 * 288; i += 128) {
    int n = i / 288, k = i % 288;
    int i2 = k / 3, r = k - i2 * 3, x = n + r - 1;
    fexpT[n * ES + k] =
        (i2 < NN && n < HH && (unsigned)x < (unsigned)HH) ? sL[i2 * XS + x] : (_Float16)0.f;
  }
  __syncthreads();

  // out = conv1x3(g_out) + cnnb; finalA += conf*out
  float cbt = sig[(size_t)b * TT + t];
  for (int tt2 = wave * 3; tt2 < wave * 3 + 3; ++tt2) {
    int mt = tt2 >> 1, nt = tt2 & 1;
    v8f acc = {0.f, 0.f, 0.f, 0.f, 0.f, 0.f, 0.f, 0.f};
    for (int k0 = 0; k0 < 288; k0 += 32) {
      v16h a = frag16([&](int m, int k) { return Wf[m * 288 + k]; }, mt * 16, k0);
      v16h bf = frag16([&](int n, int k) { return fexpT[n * ES + k]; }, nt * 16, k0);
      acc = wmma_f16(a, bf, acc);
    }
    storeC(acc, [&](int m, int n, float v) {
      if (m < NN && n < HH)
        finalA[(size_t)b * (NN * HH) + m * HH + n] += cbt * (v + cnnb[m]);
    }, mt * 16, nt * 16);
  }
}

// ---------------- MLP prep: f32->f16 A, transposed f16 B mats ----------------
__global__ void prep_mlp_kernel(const float* finalA, const float* l1w, const float* l2w,
                                _Float16* A1, _Float16* B1, _Float16* B2) {
  int gid = blockIdx.x * blockDim.x + threadIdx.x;
  int stride = gridDim.x * blockDim.x;
  for (int i = gid; i < BB * 2720; i += stride) {
    int b = i / 2720, k = i % 2720;
    A1[i] = (k < 2700) ? (_Float16)finalA[(size_t)b * 2700 + k] : (_Float16)0.f;
  }
  for (int i = gid; i < 2720 * 1024; i += stride) {
    int k = i / 1024, n = i % 1024;
    B1[i] = (k < 2700) ? (_Float16)l1w[(size_t)n * 2700 + k] : (_Float16)0.f;
  }
  for (int i = gid; i < 1024 * 256; i += stride) {
    int k = i / 256, n = i % 256;
    B2[i] = (_Float16)l2w[n * 1024 + k];
  }
}

// ---------------- generic WMMA GEMM: C(MxN) = A(f16,lda) @ B(f16,ldb) + bias ----------------
__global__ __launch_bounds__(128)
void gemm_wmma(const _Float16* __restrict__ A, int lda, const _Float16* __restrict__ B,
               int ldb, const float* __restrict__ bias, float* __restrict__ C, int N, int K) {
  __shared__ _Float16 sA[64 * 36];   // [m][k], padded stride
  __shared__ _Float16 sBT[64 * 36];  // [n][k], transposed, padded stride
  const int tid = threadIdx.x;
  const int wave = tid >> 5;
  const int bm = blockIdx.y * 64;
  const int bn = blockIdx.x * 64;
  v8f acc[4];
#pragma unroll
  for (int i = 0; i < 4; ++i) acc[i] = (v8f){0.f, 0.f, 0.f, 0.f, 0.f, 0.f, 0.f, 0.f};
  for (int k0 = 0; k0 < K; k0 += 32) {
    for (int i = tid; i < 64 * 32; i += 128)
      sA[(i >> 5) * 36 + (i & 31)] = A[(size_t)(bm + (i >> 5)) * lda + k0 + (i & 31)];
    for (int i = tid; i < 32 * 64; i += 128)
      sBT[(i & 63) * 36 + (i >> 6)] = B[(size_t)(k0 + (i >> 6)) * ldb + bn + (i & 63)];
    __syncthreads();
    v16h a = frag16([&](int m, int k) { return sA[m * 36 + k]; }, wave * 16, 0);
#pragma unroll
    for (int nt = 0; nt < 4; ++nt) {
      v16h bf = frag16([&](int n, int k) { return sBT[n * 36 + k]; }, nt * 16, 0);
      acc[nt] = wmma_f16(a, bf, acc[nt]);
    }
    __syncthreads();
  }
#pragma unroll
  for (int nt = 0; nt < 4; ++nt)
    storeC(acc[nt], [&](int m, int n, float v) {
      C[(size_t)(bm + m) * N + bn + n] = v + bias[bn + n];
    }, wave * 16, nt * 16);
}

// ---------------- batchnorm stats (biased var) -> per-feature scale/shift ----------------
__global__ __launch_bounds__(256)
void bnstats_kernel(const float* __restrict__ X, int N, const float* __restrict__ g,
                    const float* __restrict__ bta, float* scale, float* shift) {
  __shared__ float rs[256], rq[256];
  int f = blockIdx.x, tid = threadIdx.x;
  float s = 0.f, q = 0.f;
  for (int r = tid; r < BB; r += 256) {
    float x = X[(size_t)r * N + f];
    s += x; q += x * x;
  }
  rs[tid] = s; rq[tid] = q;
  __syncthreads();
  for (int o = 128; o > 0; o >>= 1) {
    if (tid < o) { rs[tid] += rs[tid + o]; rq[tid] += rq[tid + o]; }
    __syncthreads();
  }
  if (tid == 0) {
    float mean = rs[0] * (1.0f / BB);
    float var = rq[0] * (1.0f / BB) - mean * mean;
    float sc = g[f] * rsqrtf(var + 1e-5f);
    scale[f] = sc; shift[f] = bta[f] - mean * sc;
  }
}

__global__ void applybn_kernel(const float* X, const float* scale, const float* shift,
                               _Float16* Y, int total, int N) {
  int gid = blockIdx.x * blockDim.x + threadIdx.x;
  int stride = gridDim.x * blockDim.x;
  for (int i = gid; i < total; i += stride) {
    int f = i % N;
    Y[i] = (_Float16)(X[i] * scale[f] + shift[f]);
  }
}

// ---------------- bn2 + 256->2 linear + softmax, one wave per row ----------------
__global__ __launch_bounds__(128)
void final_kernel(const float* __restrict__ h2, const float* __restrict__ sc,
                  const float* __restrict__ sh, const float* __restrict__ l3w,
                  const float* __restrict__ l3b, float* __restrict__ out) {
  int row = blockIdx.x * 4 + (threadIdx.x >> 5);
  int lane = threadIdx.x & 31;
  float z0 = 0.f, z1 = 0.f;
  for (int k = lane; k < 256; k += 32) {
    float x = h2[(size_t)row * 256 + k] * sc[k] + sh[k];
    z0 += x * l3w[k];
    z1 += x * l3w[256 + k];
  }
  z0 = wave_sum(z0); z1 = wave_sum(z1);
  if (lane == 0) {
    z0 += l3b[0]; z1 += l3b[1];
    float m = fmaxf(z0, z1);
    float e0 = expf(z0 - m), e1 = expf(z1 - m);
    float d = e0 + e1;
    out[row * 2] = e0 / d;
    out[row * 2 + 1] = e1 / d;
  }
}

extern "C" void kernel_launch(void* const* d_in, const int* in_sizes, int n_in,
                              void* d_out, int out_size, void* d_ws, size_t ws_size,
                              hipStream_t stream) {
  (void)in_sizes; (void)n_in; (void)out_size; (void)ws_size;
  const float* fdata = (const float*)d_in[0];
  const float* un_w = (const float*)d_in[1];
  const float* un_b = (const float*)d_in[2];
  const float* U1   = (const float*)d_in[3];
  const float* U2   = (const float*)d_in[4];
  const float* U3   = (const float*)d_in[5];
  const float* b_e  = (const float*)d_in[6];
  const float* v_e  = (const float*)d_in[7];
  const float* gc1  = (const float*)d_in[8];
  const float* gc2  = (const float*)d_in[9];
  const float* cw   = (const float*)d_in[10];
  const float* cb   = (const float*)d_in[11];
  const float* lw   = (const float*)d_in[12];
  const float* lb   = (const float*)d_in[13];
  const float* l1w  = (const float*)d_in[14];
  const float* l1b  = (const float*)d_in[15];
  const float* bn1g = (const float*)d_in[16];
  const float* bn1b = (const float*)d_in[17];
  const float* l2w  = (const float*)d_in[18];
  const float* l2b  = (const float*)d_in[19];
  const float* bn2g = (const float*)d_in[20];
  const float* bn2b = (const float*)d_in[21];
  const float* l3w  = (const float*)d_in[22];
  const float* l3b  = (const float*)d_in[23];
  float* out = (float*)d_out;

  char* base = (char*)d_ws;
  size_t off = 0;
  auto take = [&](size_t bytes) -> char* {
    char* p = base + off;
    off += (bytes + 255) & ~(size_t)255;
    return p;
  };
  float* sig      = (float*)take(sizeof(float) * BB * TT);
  float* finalA   = (float*)take(sizeof(float) * BB * NN * HH);
  float* gmAcc    = (float*)take(sizeof(float) * 32);
  float* gl       = (float*)take(sizeof(float) * 32);
  _Float16* Wf    = (_Float16*)take(sizeof(_Float16) * 96 * 288);
  _Float16* G1t   = (_Float16*)take(sizeof(_Float16) * 32 * 32);
  _Float16* G2t   = (_Float16*)take(sizeof(_Float16) * 32 * 32);
  size_t seqBytes = sizeof(float) * (size_t)TT * BB * NN * FF;
  char* big = take(seqBytes);
  float* seq = (float*)big;
  // MLP buffers alias the seq region (seq is dead after the ST loop)
  size_t o2 = 0;
  auto take2 = [&](size_t bytes) -> char* {
    char* p = big + o2;
    o2 += (bytes + 255) & ~(size_t)255;
    return p;
  };
  _Float16* A1 = (_Float16*)take2(sizeof(_Float16) * (size_t)BB * 2720);
  _Float16* B1 = (_Float16*)take2(sizeof(_Float16) * 2720 * 1024);
  float* h1    = (float*)take2(sizeof(float) * (size_t)BB * 1024);
  float* sc1   = (float*)take2(sizeof(float) * 1024);
  float* sh1   = (float*)take2(sizeof(float) * 1024);
  _Float16* A2 = (_Float16*)take2(sizeof(_Float16) * (size_t)BB * 1024);
  _Float16* B2 = (_Float16*)take2(sizeof(_Float16) * 1024 * 256);
  float* h2    = (float*)take2(sizeof(float) * (size_t)BB * 256);
  float* sc2   = (float*)take2(sizeof(float) * 256);
  float* sh2   = (float*)take2(sizeof(float) * 256);

  uncert_kernel<<<dim3(BB * TT / 4), dim3(128), 0, stream>>>(fdata, un_w, un_b, sig);
  tatt_kernel<<<dim3(BB), dim3(128), 0, stream>>>(fdata, U1, U2, U3, b_e, v_e, seq);
  init_kernel<<<dim3(1024), dim3(256), 0, stream>>>(finalA, gmAcc, Wf, G1t, G2t, cw, gc1, gc2);
  for (int t = 0; t < TT; ++t) {
    gl_kernel<<<dim3(1), dim3(32), 0, stream>>>(gmAcc, gl, lw, lb);
    st_kernel<<<dim3(BB), dim3(128), 0, stream>>>(seq + (size_t)t * BB * NN * FF, sig, gl, cb,
                                                  Wf, G1t, G2t, finalA, gmAcc, t);
  }
  prep_mlp_kernel<<<dim3(1024), dim3(256), 0, stream>>>(finalA, l1w, l2w, A1, B1, B2);
  gemm_wmma<<<dim3(1024 / 64, BB / 64), dim3(128), 0, stream>>>(A1, 2720, B1, 1024, l1b, h1, 1024, 2720);
  bnstats_kernel<<<dim3(1024), dim3(256), 0, stream>>>(h1, 1024, bn1g, bn1b, sc1, sh1);
  applybn_kernel<<<dim3(1024), dim3(256), 0, stream>>>(h1, sc1, sh1, A2, BB * 1024, 1024);
  gemm_wmma<<<dim3(256 / 64, BB / 64), dim3(128), 0, stream>>>(A2, 1024, B2, 256, l2b, h2, 256, 1024);
  bnstats_kernel<<<dim3(256), dim3(256), 0, stream>>>(h2, 256, bn2g, bn2b, sc2, sh2);
  final_kernel<<<dim3(BB / 4), dim3(128), 0, stream>>>(h2, sc2, sh2, l3w, l3b, out);
}
